// FlashPhi3SmallAttention_23545010717553
// MI455X (gfx1250) — compile-verified, matrix operand order checked by reading, not run on hardware
//
#include <hip/hip_runtime.h>
#include <hip/hip_bf16.h>

// ---------------------------------------------------------------------------
// Phi3-small attention block for MI455X (gfx1250, wave32, WMMA).
//   hidden(4096x4096) @ qkv_w(4096x6144)+b -> RoPE -> bf16 KV cache scatter
//   -> causal GQA flash attention -> @ o_w(4096x4096) -> out f32
// All matmuls on v_wmma_f32_16x16x32_bf16; GEMMs double-buffer LDS to overlap
// HBM fetch with WMMA; flash K-tile staging uses global_load_async_to_lds_b128
// (ASYNCcnt) to bypass VGPRs.
// Workspace (bf16 bit patterns as unsigned short), 144 MiB total:
//   qkv 50,331,648 B | q 33,554,432 B | kcache 16,777,216 B
//   vcache 16,777,216 B | attn 33,554,432 B
// ---------------------------------------------------------------------------

typedef __bf16 bf16_t;
typedef bf16_t v16bf __attribute__((ext_vector_type(16)));
typedef float  v8f   __attribute__((ext_vector_type(8)));

union FragBF { v16bf v; unsigned u[8]; };

static __device__ __forceinline__ unsigned short f2bf(float f) {
    unsigned u = __builtin_bit_cast(unsigned, f);
    unsigned r = (u + 0x7FFFu + ((u >> 16) & 1u)) >> 16;   // round-to-nearest-even
    return (unsigned short)r;
}
static __device__ __forceinline__ float bf2f(unsigned short h) {
    unsigned u = ((unsigned)h) << 16;
    return __builtin_bit_cast(float, u);
}
static __device__ __forceinline__ unsigned pack2(float a, float b) {
    return (unsigned)f2bf(a) | ((unsigned)f2bf(b) << 16);
}

// Constants of this problem instance
#define TT   4096      // B*S tokens
#define SS   1024
#define BB   4
#define HH   32
#define HKV_ 8
#define DD   128
#define NQKV 6144      // (H + 2*HKV) * D
#define NMODEL 4096
#define ATT_SCALE 0.08838834764831845f   // 1/sqrt(128)

// ---------------------------------------------------------------------------
// Tiled bf16 WMMA GEMM:  C[M,N] = A[M,K] * B[K,N] (+bias)
// Block tile 128x128x32, 256 threads = 8 waves (4x2), each wave 32x64
// (2x4 WMMA tiles). Two-stage LDS double buffer: fetch chunk k+1 into
// registers while chunk k's 8 WMMAs run, store to the alternate buffer,
// one barrier per chunk.
// ---------------------------------------------------------------------------
template <bool A_BF16, bool BIAS, bool OUT_BF16>
__global__ __launch_bounds__(256)
void gemm_wmma_kernel(const void* __restrict__ Ap, const float* __restrict__ Bw,
                      const float* __restrict__ bias, void* __restrict__ Cp,
                      int M, int N, int K)
{
    __shared__ unsigned As[2][128][17];   // [buf][row m][16 u32 = 32 bf16 K] + pad
    __shared__ unsigned Bs[2][128][17];   // [buf][col n][16 u32 = 32 bf16 K] + pad

    const int tid  = threadIdx.x;
    const int bm   = blockIdx.y * 128;
    const int bn   = blockIdx.x * 128;
    const int wave = tid >> 5;
    const int lane = tid & 31;
    const int wm   = wave >> 1;          // 0..3
    const int wn   = wave & 1;           // 0..1
    const int l15  = lane & 15;
    const int koff4 = (lane >> 4) * 4;   // u32-pair offset: half-wave K split
    const int rbase = (lane >> 4) << 3;  // C-matrix row split

    const int ar  = tid >> 3;            // A stage: row base (0..31), 4 passes
    const int ac4 = (tid & 7) * 4;       // A stage: col (f32/bf16 elems)
    const int bkr = tid >> 5;            // B stage: k base (0..7), 4 passes
    const int bnc = (tid & 31) * 4;      // B stage: col

    v8f acc[2][4];
    for (int i = 0; i < 2; ++i)
        for (int j = 0; j < 4; ++j)
            for (int r = 0; r < 8; ++r) acc[i][j][r] = 0.0f;

    float4 abuf[4];
    uint2  abuf16[4];
    float4 bbuf[4];

    const int nchunks = K >> 5;

    auto global_fetch = [&](int kcc) {
        for (int pass = 0; pass < 4; ++pass) {
            const int row = ar + pass * 32;
            if (A_BF16) {
                const unsigned short* Ah = (const unsigned short*)Ap;
                abuf16[pass] = *(const uint2*)(Ah + (size_t)(bm + row) * K
                                                  + (size_t)kcc * 32 + ac4);
            } else {
                const float* Af = (const float*)Ap;
                abuf[pass] = *(const float4*)(Af + (size_t)(bm + row) * K
                                                 + (size_t)kcc * 32 + ac4);
            }
            const int k = bkr + pass * 8;
            bbuf[pass] = *(const float4*)(Bw + (size_t)(kcc * 32 + k) * N + bn + bnc);
        }
    };
    auto lds_store = [&](int buf) {
        for (int pass = 0; pass < 4; ++pass) {
            const int row = ar + pass * 32;
            if (A_BF16) {
                As[buf][row][(ac4 >> 1)]     = abuf16[pass].x;
                As[buf][row][(ac4 >> 1) + 1] = abuf16[pass].y;
            } else {
                As[buf][row][(ac4 >> 1)]     = pack2(abuf[pass].x, abuf[pass].y);
                As[buf][row][(ac4 >> 1) + 1] = pack2(abuf[pass].z, abuf[pass].w);
            }
            const int k = bkr + pass * 8;
            unsigned short* BsH = (unsigned short*)Bs[buf];
            BsH[(bnc + 0) * 34 + k] = f2bf(bbuf[pass].x);
            BsH[(bnc + 1) * 34 + k] = f2bf(bbuf[pass].y);
            BsH[(bnc + 2) * 34 + k] = f2bf(bbuf[pass].z);
            BsH[(bnc + 3) * 34 + k] = f2bf(bbuf[pass].w);
        }
    };

    // prologue: stage chunk 0
    global_fetch(0);
    lds_store(0);

    for (int kc = 0; kc < nchunks; ++kc) {
        __syncthreads();                       // buffer (kc&1) ready for all waves
        const int cur = kc & 1;
        if (kc + 1 < nchunks) {
            global_fetch(kc + 1);              // overlap HBM fetch with WMMA below
            if (kc + 2 < nchunks)
                __builtin_prefetch(Bw + (size_t)((kc + 2) * 32 + bkr) * N + bn + bnc, 0, 0);
        }

        // ---- build fragments and issue 8 WMMAs on current buffer ----
        FragBF af[2], bfg[4];
        for (int i = 0; i < 2; ++i) {
            const int row = wm * 32 + i * 16 + l15;
            for (int j = 0; j < 8; ++j) {
                const int p = (j < 4) ? (koff4 + j) : (4 + koff4 + j);
                af[i].u[j] = As[cur][row][p];
            }
        }
        for (int j4 = 0; j4 < 4; ++j4) {
            const int col = wn * 64 + j4 * 16 + l15;
            for (int j = 0; j < 8; ++j) {
                const int p = (j < 4) ? (koff4 + j) : (4 + koff4 + j);
                bfg[j4].u[j] = Bs[cur][col][p];
            }
        }
        for (int i = 0; i < 2; ++i)
            for (int j = 0; j < 4; ++j)
                acc[i][j] = __builtin_amdgcn_wmma_f32_16x16x32_bf16(
                    false, af[i].v, false, bfg[j].v, (short)0, acc[i][j], false, false);

        if (kc + 1 < nchunks)
            lds_store((kc + 1) & 1);           // alternate buffer; no extra barrier
    }

    // ---- epilogue: bias + store ----
    for (int i = 0; i < 2; ++i)
        for (int j = 0; j < 4; ++j) {
            const int col  = bn + wn * 64 + j * 16 + l15;
            const float bv = BIAS ? bias[col] : 0.0f;
            const int row0 = bm + wm * 32 + i * 16 + rbase;
            for (int r = 0; r < 8; ++r) {
                const float v = acc[i][j][r] + bv;
                if (OUT_BF16)
                    ((unsigned short*)Cp)[(size_t)(row0 + r) * N + col] = f2bf(v);
                else
                    ((float*)Cp)[(size_t)(row0 + r) * N + col] = v;
            }
        }
}

// ---------------------------------------------------------------------------
// RoPE + split + KV-cache scatter (reads bf16 qkv, writes bf16 q / kc / vc).
// One thread per rotation pair (d, d+64) of one flat head.
// ---------------------------------------------------------------------------
__global__ __launch_bounds__(256)
void rope_scatter_kernel(const unsigned short* __restrict__ qkv,
                         const float* __restrict__ cosp, const float* __restrict__ sinp,
                         const int* __restrict__ slots,
                         unsigned short* __restrict__ qout,
                         unsigned short* __restrict__ kc,
                         unsigned short* __restrict__ vc)
{
    const int e   = blockIdx.x * blockDim.x + threadIdx.x;   // T * 3072
    const int t   = e / 3072;
    const int idx = e % 3072;
    const int hf  = idx >> 6;    // flat head 0..47 (32 q, 8 k, 8 v)
    const int d   = idx & 63;

    const unsigned short* src = qkv + (size_t)t * NQKV + hf * DD;
    float x1 = bf2f(src[d]);
    float x2 = bf2f(src[d + 64]);
    float o1, o2;
    if (hf < HH + HKV_) {
        const float c = cosp[t * 64 + d];
        const float s = sinp[t * 64 + d];
        o1 = x1 * c - x2 * s;
        o2 = x1 * s + x2 * c;
    } else {
        o1 = x1; o2 = x2;
    }
    unsigned short* dst;
    if (hf < HH) {
        dst = qout + (size_t)t * (HH * DD) + hf * DD;
    } else {
        const int slot = slots[t];
        const int hk = hf - HH;
        if (hk < HKV_) dst = kc + (size_t)slot * (HKV_ * DD) + hk * DD;
        else           dst = vc + (size_t)slot * (HKV_ * DD) + (hk - HKV_) * DD;
    }
    dst[d]      = f2bf(o1);
    dst[d + 64] = f2bf(o2);
}

// ---------------------------------------------------------------------------
// Causal GQA flash attention. One wave32 per (b, h, 16-query tile).
// K tile staged via global_load_async_to_lds_b128 (ASYNCcnt, no VGPR bounce);
// V staged with a scalar transpose to [d][key]. Scores: 2x(16q x 16k) tiles
// per 32-key chunk (4 WMMAs each over D=128); online softmax with shfl_xor
// row reductions; P transits LDS into A-fragment layout; 8 PV WMMAs.
// ---------------------------------------------------------------------------
__global__ __launch_bounds__(32)
void flash_attn_kernel(const unsigned short* __restrict__ q,
                       const unsigned short* __restrict__ kcp,
                       const unsigned short* __restrict__ vcp,
                       const int* __restrict__ slots,
                       unsigned short* __restrict__ out)
{
    __shared__ __align__(16) unsigned Ks[32][68]; // [key][64 u32 d-pairs], 272B rows:
                                                  // 16B-aligned for async B128, banks 4*key+idx
    __shared__ unsigned Vs[128][17];  // [d][16 u32 key-pairs] + pad
    __shared__ unsigned Ps[16][17];   // [q-row][16 u32 key-pairs] + pad

    const int bid = blockIdx.x;
    const int qt  = bid & 63;
    const int h   = (bid >> 6) & 31;
    const int b   = bid >> 11;
    const int qb  = qt * 16;
    const int hkv = h >> 2;             // G = 4

    const int lane  = threadIdx.x;
    const int l15   = lane & 15;
    const int koff4 = (lane >> 4) * 4;
    const int rbase = (lane >> 4) << 3;

    // ---- Q fragments (4 D-chunks of 32), kept in registers for whole tile ----
    FragBF qf[4];
    {
        const unsigned* qu = (const unsigned*)q;
        const size_t base = ((size_t)(b * SS + qb + l15) * (HH * DD) + h * DD) >> 1;
        for (int dc = 0; dc < 4; ++dc)
            for (int j = 0; j < 8; ++j) {
                const int p = (j < 4) ? (koff4 + j) : (4 + koff4 + j);
                qf[dc].u[j] = qu[base + dc * 16 + p];
            }
    }

    v8f acc[8];
    float m[8], l[8];
    for (int dt = 0; dt < 8; ++dt)
        for (int r = 0; r < 8; ++r) acc[dt][r] = 0.0f;
    for (int r = 0; r < 8; ++r) { m[r] = -1e30f; l[r] = 0.0f; }

    for (int kb = 0; kb < qb + 16; kb += 32) {
        // ---- stage K tile (32 keys x 256B) via async DMA to LDS ----
#if defined(__HIP_DEVICE_COMPILE__)
        {
            const int half = lane >> 4;     // 2 rows per issue, 16B per lane
            for (int r2 = 0; r2 < 16; ++r2) {
                const int kk   = r2 * 2 + half;
                const int slot = slots[b * SS + kb + kk];
                const unsigned short* krow = kcp + (size_t)slot * (HKV_ * DD) + hkv * DD;
                unsigned long long ga = (unsigned long long)(krow + l15 * 8);
                // generic LDS pointer: low 32 bits are the LDS byte offset
                unsigned lb = (unsigned)(size_t)(&Ks[kk][0]) + (unsigned)(l15 * 16);
                asm volatile("global_load_async_to_lds_b128 %0, %1, off"
                             :: "v"(lb), "v"(ga) : "memory");
            }
            asm volatile("s_wait_asynccnt 0" ::: "memory");
        }
#else
        for (int kk = 0; kk < 32; ++kk) {
            const int slot = slots[b * SS + kb + kk];
            const unsigned* kcu = (const unsigned*)kcp;
            const size_t rb = ((size_t)slot * (HKV_ * DD) + hkv * DD) >> 1;
            Ks[kk][lane]      = kcu[rb + lane];
            Ks[kk][32 + lane] = kcu[rb + 32 + lane];
        }
#endif
        // ---- stage V tile transposed to [d][key] ----
        {
            unsigned short* VsH = (unsigned short*)Vs;
            for (int kk = 0; kk < 32; ++kk) {
                const int slot = slots[b * SS + kb + kk];
                const unsigned short* vrow = vcp + (size_t)slot * (HKV_ * DD) + hkv * DD;
                for (int i = 0; i < 4; ++i) {
                    const int d = lane * 4 + i;
                    VsH[d * 34 + kk] = vrow[d];
                }
            }
        }
        __syncthreads();

        // ---- scores: two 16x16 tiles, K-dim = D in 4 chunks ----
        v8f sc[2];
        for (int st = 0; st < 2; ++st)
            for (int r = 0; r < 8; ++r) sc[st][r] = 0.0f;
        for (int st = 0; st < 2; ++st) {
            const int key = st * 16 + l15;
            for (int dc = 0; dc < 4; ++dc) {
                FragBF kf;
                for (int j = 0; j < 8; ++j) {
                    const int p = (j < 4) ? (koff4 + j) : (4 + koff4 + j);
                    kf.u[j] = Ks[key][dc * 16 + p];
                }
                sc[st] = __builtin_amdgcn_wmma_f32_16x16x32_bf16(
                    false, qf[dc].v, false, kf.v, (short)0, sc[st], false, false);
            }
        }

        // ---- scale + causal mask ----
        for (int st = 0; st < 2; ++st)
            for (int r = 0; r < 8; ++r) {
                const int key = kb + st * 16 + l15;
                const int row = qb + rbase + r;
                const float s = sc[st][r] * ATT_SCALE;
                sc[st][r] = (key <= row) ? s : -1e30f;
            }

        // ---- online softmax (row reductions across 16-lane halves) ----
        for (int r = 0; r < 8; ++r) {
            float mx = fmaxf(sc[0][r], sc[1][r]);
            for (int off = 1; off < 16; off <<= 1)
                mx = fmaxf(mx, __shfl_xor(mx, off, 32));
            const float mn   = fmaxf(m[r], mx);
            const float corr = __expf(m[r] - mn);
            m[r] = mn;
            const float p0 = __expf(sc[0][r] - mn);
            const float p1 = __expf(sc[1][r] - mn);
            sc[0][r] = p0; sc[1][r] = p1;
            float psum = p0 + p1;
            for (int off = 1; off < 16; off <<= 1)
                psum += __shfl_xor(psum, off, 32);
            l[r] = l[r] * corr + psum;
            for (int dt = 0; dt < 8; ++dt) acc[dt][r] *= corr;
        }

        // ---- transpose P through LDS into A-fragment layout ----
        unsigned short* PsH = (unsigned short*)Ps;
        for (int st = 0; st < 2; ++st)
            for (int r = 0; r < 8; ++r)
                PsH[(rbase + r) * 34 + st * 16 + l15] = f2bf(sc[st][r]);
        __syncthreads();

        FragBF pf;
        for (int j = 0; j < 8; ++j) {
            const int p = (j < 4) ? (koff4 + j) : (4 + koff4 + j);
            pf.u[j] = Ps[l15][p];
        }
        // ---- PV: 8 d-tiles of 16, K-dim = 32 keys ----
        for (int dt = 0; dt < 8; ++dt) {
            FragBF vf;
            const int d = dt * 16 + l15;
            for (int j = 0; j < 8; ++j) {
                const int p = (j < 4) ? (koff4 + j) : (4 + koff4 + j);
                vf.u[j] = Vs[d][p];
            }
            acc[dt] = __builtin_amdgcn_wmma_f32_16x16x32_bf16(
                false, pf.v, false, vf.v, (short)0, acc[dt], false, false);
        }
        __syncthreads();   // before next chunk overwrites Ks/Vs
    }

    // ---- normalize and store bf16 attention output ----
    for (int dt = 0; dt < 8; ++dt)
        for (int r = 0; r < 8; ++r) {
            const int row = b * SS + qb + rbase + r;
            const int col = h * DD + dt * 16 + l15;
            out[(size_t)row * (HH * DD) + col] = f2bf(acc[dt][r] / l[r]);
        }
}

// ---------------------------------------------------------------------------
extern "C" void kernel_launch(void* const* d_in, const int* in_sizes, int n_in,
                              void* d_out, int out_size, void* d_ws, size_t ws_size,
                              hipStream_t stream)
{
    (void)in_sizes; (void)n_in; (void)out_size; (void)ws_size;

    const float* hidden = (const float*)d_in[0];
    const float* cosp   = (const float*)d_in[1];
    const float* sinp   = (const float*)d_in[2];
    const float* qkv_w  = (const float*)d_in[3];
    const float* qkv_b  = (const float*)d_in[4];
    const float* o_w    = (const float*)d_in[5];
    // d_in[6]/d_in[7]: f32 k/v caches (inputs; must not be mutated — we keep
    // private bf16 caches in workspace instead)
    const int* slots    = (const int*)d_in[8];

    char* ws = (char*)d_ws;
    unsigned short* qkv_bf  = (unsigned short*)(ws);
    unsigned short* q_bf    = (unsigned short*)(ws + 50331648ull);
    unsigned short* kcache  = (unsigned short*)(ws + 50331648ull + 33554432ull);
    unsigned short* vcache  = (unsigned short*)(ws + 50331648ull + 33554432ull + 16777216ull);
    unsigned short* attn_bf = (unsigned short*)(ws + 50331648ull + 33554432ull + 2ull * 16777216ull);

    // 1) QKV projection + bias -> bf16 qkv
    dim3 g1(NQKV / 128, TT / 128);
    gemm_wmma_kernel<false, true, true><<<g1, dim3(256), 0, stream>>>(
        hidden, qkv_w, qkv_b, qkv_bf, TT, NQKV, NMODEL);

    // 2) RoPE + split + KV cache scatter
    const int nrope = TT * 3072;
    rope_scatter_kernel<<<nrope / 256, dim3(256), 0, stream>>>(
        qkv_bf, cosp, sinp, slots, q_bf, kcache, vcache);

    // 3) Flash attention: one wave per (b, h, 16-query tile)
    flash_attn_kernel<<<BB * HH * (SS / 16), dim3(32), 0, stream>>>(
        q_bf, kcache, vcache, slots, attn_bf);

    // 4) Output projection -> f32 d_out
    dim3 g3(NMODEL / 128, TT / 128);
    gemm_wmma_kernel<true, false, false><<<g3, dim3(256), 0, stream>>>(
        attn_bf, o_w, nullptr, d_out, TT, NMODEL, NMODEL);
}
